// RNN_66133906424538
// MI455X (gfx1250) — matrix operand order repaired and checked by reading, not compiled, not run on hardware
//
#include <hip/hip_runtime.h>

// ---------------------------------------------------------------------------
// CDNA5 (gfx1250) 2-layer Elman RNN, software-pipelined persistent scan.
//   Phase 1: Z0 = x @ Wih0^T + (b_ih0 + b_hh0)            (parallel WMMA GEMM)
//   Phase 2: one persistent kernel, 48 WGs in 3 pipelined groups:
//     role 0 (WG  0-15): layer-0 scan  h_{t+1}=tanh(Z0[t]+h_t@Whh0^T) -> h0 ring
//     role 1 (WG 16-31): Z1[t] = y0[t] @ Wih1^T + b1  -> Z1 ring (no recurrence)
//     role 2 (WG 32-47): layer-1 scan  -> y1 (d_out), h_n
// All matmuls via v_wmma_f32_16x16x32_bf16 (fp32 accumulate).
// Pipeline sync: monotonic per-stage counters, 8-deep rings with back-pressure.
// ---------------------------------------------------------------------------

typedef __attribute__((ext_vector_type(16))) __bf16 v16bf;
typedef __attribute__((ext_vector_type(8)))  __bf16 v8bf;
typedef __attribute__((ext_vector_type(8)))  float  v8f;

union V16 { v16bf v; v8bf h[2]; };

static __device__ __forceinline__ v8f wmma_bf16(v16bf a, v16bf b, v8f c) {
  return __builtin_amdgcn_wmma_f32_16x16x32_bf16(false, a, false, b,
                                                 (short)0, c, false, false);
}

// B stationary slice: B[k][n] = W[n][k]; lane chunk K = kc*32 + half*16 + e
static __device__ __forceinline__ void loadBreg(const __bf16* __restrict__ Wbf,
                                                int n, int half, v16bf* Breg) {
#pragma unroll
  for (int kc = 0; kc < 16; ++kc) {
    V16 b;
    const __bf16* bp = Wbf + (size_t)n * 512 + kc * 32 + half * 16;
    b.h[0] = *(const v8bf*)bp;
    b.h[1] = *(const v8bf*)(bp + 8);
    Breg[kc] = b.v;
  }
}

// 16 chained WMMAs over K=512; A row base already offset by am*512.
static __device__ __forceinline__ v8f mm512(const __bf16* __restrict__ Abase,
                                            int half, const v16bf* Breg, v8f c) {
#pragma unroll
  for (int kc = 0; kc < 16; ++kc) {
    V16 a;
    const __bf16* ap = Abase + kc * 32 + half * 16;
    a.h[0] = *(const v8bf*)ap;
    a.h[1] = *(const v8bf*)(ap + 8);
    c = wmma_bf16(a.v, Breg[kc], c);
  }
  return c;
}

static __device__ __forceinline__ void spin_ge(const unsigned* c, unsigned target) {
  while (__hip_atomic_load(c, __ATOMIC_ACQUIRE, __HIP_MEMORY_SCOPE_AGENT) < target)
    __builtin_amdgcn_s_sleep(1);
}
static __device__ __forceinline__ void arrive(unsigned* c) {
  __threadfence();
  __hip_atomic_fetch_add(c, 1u, __ATOMIC_RELEASE, __HIP_MEMORY_SCOPE_AGENT);
}

// ---------------------------------------------------------------------------
// Prep: bf16 weight conversions, bias folding, h init, counter reset.
// grid = 1024 x 256 = 262144 = 512*512
// ---------------------------------------------------------------------------
__global__ void __launch_bounds__(256)
rnn_prep(const float* __restrict__ Whh0, const float* __restrict__ Wih1,
         const float* __restrict__ Whh1,
         const float* __restrict__ bih0, const float* __restrict__ bhh0,
         const float* __restrict__ bih1, const float* __restrict__ bhh1,
         const float* __restrict__ hx,
         __bf16* __restrict__ whh0bf, __bf16* __restrict__ wih1bf,
         __bf16* __restrict__ whh1bf,
         float* __restrict__ b0, float* __restrict__ b1,
         __bf16* __restrict__ h0ring, __bf16* __restrict__ h1A,
         unsigned* __restrict__ cnts) {
  const int i = blockIdx.x * 256 + threadIdx.x;
  if (i < 512 * 512) {
    whh0bf[i] = (__bf16)Whh0[i];
    wih1bf[i] = (__bf16)Wih1[i];
    whh1bf[i] = (__bf16)Whh1[i];
  }
  if (i < 512) {
    b0[i] = bih0[i] + bhh0[i];
    b1[i] = bih1[i] + bhh1[i];
  }
  if (i < 64 * 512) {
    h0ring[i] = (__bf16)hx[i];           // ring slot 0 = h_0
    h1A[i]    = (__bf16)hx[64 * 512 + i];
  }
  if (i < 64) cnts[i] = 0u;
}

// ---------------------------------------------------------------------------
// Parallel GEMM: Z[m][n] = sum_k src[m][k] * W[n][k] + bias[n]
// M = 65536, N = K = 512. Block tile 64x64, K-chunk 32, LDS-staged bf16.
// ---------------------------------------------------------------------------
__global__ void __launch_bounds__(256)
rnn_gemm(const float* __restrict__ src, const float* __restrict__ W,
         const float* __restrict__ bias, float* __restrict__ Z) {
  __shared__ __bf16 Alds[64 * 40];
  __shared__ __bf16 Blds[64 * 40];

  const int tid  = threadIdx.x;
  const int bm   = blockIdx.x;
  const int bn   = blockIdx.y;
  const int row  = tid >> 2;
  const int colb = (tid & 3) * 8;
  const int wave = tid >> 5, lane = tid & 31;
  const int half = lane >> 4, lr = lane & 15;
  const int mi   = wave & 3;
  const int nb2  = (wave >> 2) * 2;

  v8f acc[2] = {};

  const float* __restrict__ aSrc = src + (size_t)(bm * 64 + row) * 512;
  const float* __restrict__ bSrc = W   + (size_t)(bn * 64 + row) * 512;

  for (int kc = 0; kc < 16; ++kc) {
    {
      const float* p = aSrc + kc * 32 + colb;
      const float* q = bSrc + kc * 32 + colb;
      v8bf ta, tb;
#pragma unroll
      for (int j = 0; j < 8; ++j) {
        ta[j] = (__bf16)p[j];
        tb[j] = (__bf16)q[j];
      }
      *(v8bf*)&Alds[row * 40 + colb] = ta;
      *(v8bf*)&Blds[row * 40 + colb] = tb;
    }
    __syncthreads();

    V16 a, bv0, bv1;
    const __bf16* ap = &Alds[(mi * 16 + lr) * 40 + half * 16];
    a.h[0] = *(const v8bf*)ap;
    a.h[1] = *(const v8bf*)(ap + 8);
    const __bf16* bp0 = &Blds[((nb2 + 0) * 16 + lr) * 40 + half * 16];
    bv0.h[0] = *(const v8bf*)bp0;
    bv0.h[1] = *(const v8bf*)(bp0 + 8);
    const __bf16* bp1 = &Blds[((nb2 + 1) * 16 + lr) * 40 + half * 16];
    bv1.h[0] = *(const v8bf*)bp1;
    bv1.h[1] = *(const v8bf*)(bp1 + 8);

    acc[0] = wmma_bf16(a.v, bv0.v, acc[0]);
    acc[1] = wmma_bf16(a.v, bv1.v, acc[1]);
    __syncthreads();
  }

#pragma unroll
  for (int j = 0; j < 2; ++j) {
    const int n = bn * 64 + (nb2 + j) * 16 + lr;
    const float bvl = bias[n];
#pragma unroll
    for (int r = 0; r < 8; ++r) {
      const size_t m = (size_t)bm * 64 + mi * 16 + half * 8 + r;
      Z[m * 512 + n] = acc[j][r] + bvl;
    }
  }
}

// ---------------------------------------------------------------------------
// Persistent pipelined scan: 48 WGs x 256 threads (8 waves each).
// Each wave owns one 16x16 output tile; its B-matrix slice (512x16 bf16 =
// 128 VGPRs) stays register-resident for all 1024 steps.
// h0 ring slot j holds h_j&7 ; y0[t] == h_{t+1} == slot (t+1)&7.
// cnts[0]=cnt0 (role0 steps), cnts[16]=cntZ (role1), cnts[32]=cnt1 (role2).
// ---------------------------------------------------------------------------
__global__ void __launch_bounds__(256)
rnn_pipe(const float* __restrict__ Z0,
         const __bf16* __restrict__ whh0, const __bf16* __restrict__ wih1,
         const __bf16* __restrict__ whh1, const float* __restrict__ b1,
         __bf16* __restrict__ h0ring,                       // 8 * 64*512 bf16
         __bf16* __restrict__ h1A, __bf16* __restrict__ h1B,
         float* __restrict__ z1ring,                        // 8 * 64*512 f32
         float* __restrict__ yout, float* __restrict__ hn0,
         float* __restrict__ hn1, unsigned* __restrict__ cnts) {
  const int role = blockIdx.x >> 4;    // 0,1,2
  const int wid  = blockIdx.x & 15;
  const int tid  = threadIdx.x;
  const int wave = tid >> 5, lane = tid & 31;
  const int half = lane >> 4, lr = lane & 15;
  const int mi   = wave & 3, ni = wave >> 2;
  const int n    = wid * 32 + ni * 16 + lr;   // owned output column
  const int am   = mi * 16 + lr;              // A row read by this lane
  const int mrow = mi * 16 + half * 8;        // C row base (elem r -> mrow+r)

  unsigned* cnt0 = cnts + 0;
  unsigned* cntZ = cnts + 16;
  unsigned* cnt1 = cnts + 32;

  v16bf Breg[16];
  const size_t HSL = (size_t)64 * 512;        // one h / z slot

  if (role == 0) {
    // -------- layer-0 scan --------
    loadBreg(whh0, n, half, Breg);
    for (int t = 0; t < 1024; ++t) {
      const __bf16* hread  = h0ring + (size_t)(t & 7) * HSL;
      __bf16*       hwrite = h0ring + (size_t)((t + 1) & 7) * HSL;

      const float* zb = Z0 + ((size_t)t * 64 + mrow) * 512 + n;
      v8f c;
#pragma unroll
      for (int r = 0; r < 8; ++r) c[r] = zb[(size_t)r * 512];

      c = mm512(hread + am * 512, half, Breg, c);

#pragma unroll
      for (int r = 0; r < 8; ++r) {
        const float hv = tanhf(c[r]);
        hwrite[(mrow + r) * 512 + n] = (__bf16)hv;
        if (t == 1023) hn0[(mrow + r) * 512 + n] = hv;
      }

      __syncthreads();
      if (tid == 0) {
        arrive(cnt0);
        spin_ge(cnt0, (unsigned)(t + 1) * 16u);            // group barrier
        if (t >= 7) spin_ge(cntZ, (unsigned)(t - 6) * 16u); // ring back-pressure
      }
      __syncthreads();
    }
  } else if (role == 1) {
    // -------- Z1 producer: Z1[t] = y0[t] @ Wih1^T + b1 --------
    loadBreg(wih1, n, half, Breg);
    const float bias = b1[n];
    for (int t = 0; t < 1024; ++t) {
      if (tid == 0) {
        spin_ge(cnt0, (unsigned)(t + 1) * 16u);             // y0[t] ready
        if (t >= 8) spin_ge(cnt1, (unsigned)(t - 7) * 16u); // z1 slot free
      }
      __syncthreads();

      const __bf16* y0 = h0ring + (size_t)((t + 1) & 7) * HSL;
      v8f c;
#pragma unroll
      for (int r = 0; r < 8; ++r) c[r] = bias;

      c = mm512(y0 + am * 512, half, Breg, c);

      float* z1 = z1ring + (size_t)(t & 7) * HSL;
#pragma unroll
      for (int r = 0; r < 8; ++r) z1[(mrow + r) * 512 + n] = c[r];

      __syncthreads();
      if (tid == 0) arrive(cntZ);   // no intra-group dependency
    }
  } else {
    // -------- layer-1 scan --------
    loadBreg(whh1, n, half, Breg);
    for (int t = 0; t < 1024; ++t) {
      if (tid == 0) spin_ge(cntZ, (unsigned)(t + 1) * 16u); // Z1[t] ready
      __syncthreads();

      const __bf16* hread  = (t & 1) ? h1B : h1A;
      __bf16*       hwrite = (t & 1) ? h1A : h1B;

      const float* zb = z1ring + (size_t)(t & 7) * HSL + (size_t)mrow * 512 + n;
      v8f c;
#pragma unroll
      for (int r = 0; r < 8; ++r) c[r] = zb[(size_t)r * 512];

      c = mm512(hread + am * 512, half, Breg, c);

#pragma unroll
      for (int r = 0; r < 8; ++r) {
        const float hv = tanhf(c[r]);
        hwrite[(mrow + r) * 512 + n] = (__bf16)hv;
        yout[((size_t)t * 64 + mrow + r) * 512 + n] = hv;
        if (t == 1023) hn1[(mrow + r) * 512 + n] = hv;
      }

      __syncthreads();
      if (tid == 0) {
        arrive(cnt1);
        spin_ge(cnt1, (unsigned)(t + 1) * 16u);             // group barrier
      }
      __syncthreads();
    }
  }
}

// ---------------------------------------------------------------------------
extern "C" void kernel_launch(void* const* d_in, const int* in_sizes, int n_in,
                              void* d_out, int out_size, void* d_ws, size_t ws_size,
                              hipStream_t stream) {
  const float* x    = (const float*)d_in[0];
  const float* hx   = (const float*)d_in[1];
  const float* Wih0 = (const float*)d_in[2];
  const float* Whh0 = (const float*)d_in[3];
  const float* bih0 = (const float*)d_in[4];
  const float* bhh0 = (const float*)d_in[5];
  const float* Wih1 = (const float*)d_in[6];
  const float* Whh1 = (const float*)d_in[7];
  const float* bih1 = (const float*)d_in[8];
  const float* bhh1 = (const float*)d_in[9];
  float* out = (float*)d_out;

  // workspace layout (~131 MB)
  char* w = (char*)d_ws;
  size_t off = 0;
  float*  Z0     = (float*)(w + off);  off += (size_t)1024 * 64 * 512 * 4; // 128 MB
  __bf16* whh0bf = (__bf16*)(w + off); off += (size_t)512 * 512 * 2;
  __bf16* wih1bf = (__bf16*)(w + off); off += (size_t)512 * 512 * 2;
  __bf16* whh1bf = (__bf16*)(w + off); off += (size_t)512 * 512 * 2;
  float*  b0     = (float*)(w + off);  off += 512 * 4;
  float*  b1     = (float*)(w + off);  off += 512 * 4;
  __bf16* h0ring = (__bf16*)(w + off); off += (size_t)8 * 64 * 512 * 2;    // 512 KB
  __bf16* h1A    = (__bf16*)(w + off); off += (size_t)64 * 512 * 2;
  __bf16* h1B    = (__bf16*)(w + off); off += (size_t)64 * 512 * 2;
  float*  z1ring = (float*)(w + off);  off += (size_t)8 * 64 * 512 * 4;    // 1 MB
  unsigned* cnts = (unsigned*)(w + off); off += 256;

  float* y1  = out;                            // (S,B,H)
  float* hn0 = out + (size_t)1024 * 64 * 512;  // h_n[0]
  float* hn1 = hn0 + 64 * 512;                 // h_n[1]

  rnn_prep<<<1024, 256, 0, stream>>>(Whh0, Wih1, Whh1, bih0, bhh0, bih1, bhh1, hx,
                                     whh0bf, wih1bf, whh1bf, b0, b1,
                                     h0ring, h1A, cnts);

  // Phase 1: Z0 = x @ Wih0^T + b0 (fully parallel)
  dim3 gg(1024, 8);
  rnn_gemm<<<gg, 256, 0, stream>>>(x, Wih0, b0, Z0);

  // Phase 2: 3-stage software pipeline (both layers + layer-1 input GEMM)
  rnn_pipe<<<48, 256, 0, stream>>>(Z0, whh0bf, wih1bf, whh1bf, b1,
                                   h0ring, h1A, h1B, z1ring,
                                   y1, hn0, hn1, cnts);
}
